// GraphTrans_85916525789391
// MI455X (gfx1250) — matrix operand order repaired and checked by reading, not compiled, not run on hardware
//
#include <hip/hip_runtime.h>
#include <hip/hip_bf16.h>
#include <math.h>

// ---------------- problem constants (match reference) ----------------
#define Bg    16
#define NPG   4096
#define Nn    (Bg*NPG)        // 65536 nodes
#define DEG   8
#define Ee    (Nn*DEG)        // 524288 edges
#define GXc   15
#define GYc   15
#define NVOX  (GXc*GYc)       // 225 voxels/graph after pool5
#define NCc   (Bg*NVOX)       // 3600 pooled nodes
#define NOUTc 101
#define MAXHC 384
#define EPSbn 1e-5f
#define ENC_NEG_INF 0x007FFFFFu   // enc_f32(-inf)

// ---------------- helpers ----------------
__device__ __forceinline__ unsigned int enc_f32(float f) {
  unsigned int u = __float_as_uint(f);
  return (u & 0x80000000u) ? ~u : (u | 0x80000000u);  // order-preserving map
}
__device__ __forceinline__ float dec_f32(unsigned int u) {
  return (u & 0x80000000u) ? __uint_as_float(u & 0x7FFFFFFFu) : __uint_as_float(~u);
}
__device__ __forceinline__ bool finitef(float v) { return (v == v) && (fabsf(v) <= 3.3e38f); }
__device__ __forceinline__ float eluf(float v) { return v > 0.f ? v : (__expf(v) - 1.f); }

// ---------------- WMMA GEMM: Y[M,Out] = X[M,K] @ W[K,Out] (+bias) ----------------
typedef __attribute__((ext_vector_type(16))) _Float16 v16h;
typedef __attribute__((ext_vector_type(8)))  float    v8f;

// Block (128 threads, 4 waves) computes a 32x64 tile of Y.
// Per 32-wide k-step: branch-free cooperative staging of A(32x32), B(32x64)
// into LDS as f16. Out-of-range slots are handled by CLAMPED addresses
// (always in-bounds, 16B aligned) + zero-select -- no exec-mask branching.
// Requires M%16==0, Out%16==0, K%4==0, Out%4==0 (all call sites satisfy;
// with these, any float4 slot is fully in-range or fully out-of-range).
// Each wave reads fragments from LDS per ISA 7.12.2 layouts and issues two
// v_wmma_f32_16x16x32_f16 (row-tiles 0-15 / 16-31 share the B fragment).
__global__ __launch_bounds__(128) void k_gemm_wmma(
    const float* __restrict__ X, const float* __restrict__ W,
    const float* __restrict__ bias, float* __restrict__ Y,
    int M, int K, int Out)
{
  __shared__ _Float16 sA[32 * 32];
  __shared__ _Float16 sB[32 * 64];

  const int tid  = threadIdx.x;
  const int wave = tid >> 5;
  const int lane = tid & 31;
  const int row0 = blockIdx.x * 32;
  const int col0 = blockIdx.y * 64;

  const int l15   = lane & 15;
  const int koffA = (lane & 16) ? 8 : 0;     // A 16x32 f16 fragment layout
  const int koffB = (lane & 16) ? 8 : 0;     // B 32x16 f16 fragment layout
  const int colW  = col0 + wave * 16;        // this wave's 16 output columns
  const bool waveActive = (colW < Out);
  const bool tile1      = (row0 + 16 < M);   // second row-tile fully valid (M%16==0)

  v8f acc0 = {}, acc1 = {};

  for (int kk = 0; kk < K; kk += 32) {
    // ---- stage A: rows row0..row0+31, cols kk..kk+31 -> sA (zero padded) ----
#pragma unroll
    for (int p = 0; p < 2; ++p) {
      int idx = tid + p * 128;               // 256 float4 slots
      int r   = idx >> 3;                    // 0..31
      int c4  = (idx & 7) * 4;               // 0,4,..,28
      int gr  = row0 + r;
      int grc = min(gr, M - 1);              // clamped, always in-bounds
      int kc  = min(kk + c4, K - 4);         // clamped, 16B aligned (K%4==0)
      float4 v = *reinterpret_cast<const float4*>(X + (size_t)grc * K + kc);
      bool ok = (gr < M) && (kk + c4 + 4 <= K);  // slot fully in or fully out
      _Float16* d = &sA[r * 32 + c4];
      d[0] = (_Float16)(ok ? v.x : 0.f);
      d[1] = (_Float16)(ok ? v.y : 0.f);
      d[2] = (_Float16)(ok ? v.z : 0.f);
      d[3] = (_Float16)(ok ? v.w : 0.f);
    }
    // ---- stage B: rows kk..kk+31, cols col0..col0+63 -> sB (zero padded) ----
#pragma unroll
    for (int p = 0; p < 4; ++p) {
      int idx = tid + p * 128;               // 512 float4 slots
      int kr  = idx >> 4;                    // 0..31
      int c4  = (idx & 15) * 4;              // 0,4,..,60
      int krc = min(kk + kr, K - 1);         // clamped row
      int cc  = min(col0 + c4, Out - 4);     // clamped col, 16B aligned (Out%4==0)
      float4 v = *reinterpret_cast<const float4*>(W + (size_t)krc * Out + cc);
      bool ok = ((kk + kr) < K) && (col0 + c4 + 4 <= Out);
      _Float16* d = &sB[kr * 64 + c4];
      d[0] = (_Float16)(ok ? v.x : 0.f);
      d[1] = (_Float16)(ok ? v.y : 0.f);
      d[2] = (_Float16)(ok ? v.z : 0.f);
      d[3] = (_Float16)(ok ? v.w : 0.f);
    }
    // prefetch next k-step tiles into cache while we compute
    if (kk + 32 < K) {
      int gr = min(row0 + (tid >> 3), M - 1);
      __builtin_prefetch(X + (size_t)gr * K + min(kk + 32 + (tid & 7) * 4, K - 4), 0, 0);
      int krn = min(kk + 32 + (tid >> 4), K - 1);
      __builtin_prefetch(W + (size_t)krn * Out + min(col0 + (tid & 15) * 4, Out - 4), 0, 0);
    }
    __syncthreads();

    if (waveActive) {  // wave-uniform: EXEC all ones inside
      v16h a0, a1, b;
#pragma unroll
      for (int i = 0; i < 8; ++i) {
        int kb = koffA + ((i < 4) ? (2 * i) : (16 + 2 * (i - 4)));
        a0[2 * i]     = sA[l15 * 32 + kb];
        a0[2 * i + 1] = sA[l15 * 32 + kb + 1];
        a1[2 * i]     = sA[(l15 + 16) * 32 + kb];
        a1[2 * i + 1] = sA[(l15 + 16) * 32 + kb + 1];
      }
#pragma unroll
      for (int i = 0; i < 16; ++i)
        b[i] = sB[(koffB + ((i < 8) ? i : (i + 8))) * 64 + wave * 16 + l15];
      acc0 = __builtin_amdgcn_wmma_f32_16x16x32_f16(false, a0, false, b, (short)0, acc0,
                                                    false, false);
      acc1 = __builtin_amdgcn_wmma_f32_16x16x32_f16(false, a1, false, b, (short)0, acc1,
                                                    false, false);
    }
    __syncthreads();
  }

  if (waveActive) {
    const int nB = colW + l15;
    const int mD = ((lane & 16) ? 8 : 0);    // C/D layout: VGPR r -> M=r(+8)
    const float bv = bias ? bias[nB] : 0.0f;
#pragma unroll
    for (int r = 0; r < 8; ++r)
      Y[(size_t)(row0 + mD + r) * Out + nB] = acc0[r] + bv;
    if (tile1) {
#pragma unroll
      for (int r = 0; r < 8; ++r)
        Y[(size_t)(row0 + 16 + mD + r) * Out + nB] = acc1[r] + bv;
    }
  }
}

// scalar GEMM for K<32 or Out not multiple of 16 (conv1 K=1, FC Out=101)
__global__ void k_gemm_scalar(const float* __restrict__ X, const float* __restrict__ W,
                              const float* __restrict__ bias, float* __restrict__ Y,
                              int M, int K, int Out)
{
  int t = blockIdx.x * blockDim.x + threadIdx.x;
  if (t >= M * Out) return;
  int m = t / Out, o = t - m * Out;
  float acc = bias ? bias[o] : 0.0f;
  for (int k = 0; k < K; ++k) acc += X[(size_t)m * K + k] * W[(size_t)k * Out + o];
  Y[t] = acc;
}

// ---------------- fills ----------------
__global__ void k_fill_f(float* p, float v, size_t n) {
  size_t t = (size_t)blockIdx.x * blockDim.x + threadIdx.x;
  if (t < n) p[t] = v;
}
__global__ void k_fill_u(unsigned int* p, unsigned int v, size_t n) {
  size_t t = (size_t)blockIdx.x * blockDim.x + threadIdx.x;
  if (t < n) p[t] = v;
}
__global__ void k_fill_i(int* p, int v, size_t n) {
  size_t t = (size_t)blockIdx.x * blockDim.x + threadIdx.x;
  if (t < n) p[t] = v;
}
__global__ void k_add(float* a, const float* b, size_t n) {
  size_t t = (size_t)blockIdx.x * blockDim.x + threadIdx.x;
  if (t < n) a[t] += b[t];
}

// ---------------- edge pass 1: alpha + segment-max ----------------
// e_hc recomputed on the fly: e = edge_attr[E,3] @ We[3,HC], We staged in LDS.
__global__ void k_edge_alpha(const float* __restrict__ q, const float* __restrict__ k,
                             const float* __restrict__ We, const float* __restrict__ eattr,
                             const int* __restrict__ src, const int* __restrict__ dst,
                             const int* __restrict__ emask,
                             float* __restrict__ alpha, unsigned int* __restrict__ mmax,
                             int E_, int H, int C)
{
  __shared__ float sWe[3 * MAXHC];
  const int HC = H * C;
  for (int i = threadIdx.x; i < 3 * HC; i += blockDim.x) sWe[i] = We[i];
  __syncthreads();
  int t = blockIdx.x * blockDim.x + threadIdx.x;
  if (t >= E_ * H) return;
  int e = t / H, h = t - e * H;
  if (emask && !emask[e]) return;
  int s = src[e], d = dst[e];
  float ea0 = eattr[3 * e], ea1 = eattr[3 * e + 1], ea2 = eattr[3 * e + 2];
  const float* qd = q + (size_t)d * HC + h * C;
  const float* ks = k + (size_t)s * HC + h * C;
  float acc = 0.f;
  for (int c = 0; c < C; ++c) {
    int hc = h * C + c;
    float ehc = ea0 * sWe[hc] + ea1 * sWe[HC + hc] + ea2 * sWe[2 * HC + hc];
    acc += qd[c] * (ks[c] + ehc);
  }
  acc *= rsqrtf((float)C);
  alpha[t] = acc;
  atomicMax(&mmax[(size_t)d * H + h], enc_f32(acc));
}

// ---------------- edge pass 2: exp + denom ----------------
__global__ void k_edge_softmax(const float* __restrict__ alpha,
                               const unsigned int* __restrict__ mmax,
                               const int* __restrict__ dst, const int* __restrict__ emask,
                               float* __restrict__ aexp, float* __restrict__ denom,
                               int E_, int H)
{
  int t = blockIdx.x * blockDim.x + threadIdx.x;
  if (t >= E_ * H) return;
  int e = t / H, h = t - e * H;
  if (emask && !emask[e]) { aexp[t] = 0.f; return; }
  int d = dst[e];
  float m = dec_f32(mmax[(size_t)d * H + h]);
  if (!finitef(m)) m = 0.f;                       // reference m_safe
  float a = __expf(alpha[t] - m);
  aexp[t] = a;
  atomicAdd(&denom[(size_t)d * H + h], a);
}

// ---------------- edge pass 3: weighted scatter of (v[src]+e) ----------------
__global__ void k_edge_scatter(const float* __restrict__ v, const float* __restrict__ We,
                               const float* __restrict__ eattr,
                               const int* __restrict__ src, const int* __restrict__ dst,
                               const int* __restrict__ emask,
                               const float* __restrict__ aexp, const float* __restrict__ denom,
                               float* __restrict__ out, int E_, int H, int C)
{
  __shared__ float sWe[3 * MAXHC];
  const int HC = H * C;
  for (int i = threadIdx.x; i < 3 * HC; i += blockDim.x) sWe[i] = We[i];
  __syncthreads();
  int t = blockIdx.x * blockDim.x + threadIdx.x;
  if (t >= E_ * H) return;
  int e = t / H, h = t - e * H;
  if (emask && !emask[e]) return;
  int s = src[e], d = dst[e];
  float den = denom[(size_t)d * H + h];
  den = den > 0.f ? den : 1.f;
  float w = aexp[t] / den;
  float ea0 = eattr[3 * e], ea1 = eattr[3 * e + 1], ea2 = eattr[3 * e + 2];
  const float* vs = v + (size_t)s * HC + h * C;
  float* od = out + (size_t)d * HC + h * C;
  for (int c = 0; c < C; ++c) {
    int hc = h * C + c;
    float ehc = ea0 * sWe[hc] + ea1 * sWe[HC + hc] + ea2 * sWe[2 * HC + hc];
    atomicAdd(&od[c], w * (vs[c] + ehc));
  }
}

// ---------------- ELU + BatchNorm (training-mode batch stats) ----------------
// one block per channel: LDS tree-reduce sum / sumsq / count over rows (ELU applied)
__global__ __launch_bounds__(256) void k_bn_stats(const float* __restrict__ X,
                                                  const int* __restrict__ nmask,
                                                  float* __restrict__ stats,
                                                  int M, int HC)
{
  __shared__ float s_sum[256], s_sq[256], s_cnt[256];
  const int c = blockIdx.x;
  float sum = 0.f, sq = 0.f, cnt = 0.f;
  for (int r = threadIdx.x; r < M; r += blockDim.x) {
    if (nmask && !nmask[r]) continue;
    float v = eluf(X[(size_t)r * HC + c]);
    sum += v; sq += v * v; cnt += 1.f;
  }
  s_sum[threadIdx.x] = sum; s_sq[threadIdx.x] = sq; s_cnt[threadIdx.x] = cnt;
  __syncthreads();
  for (int o = blockDim.x >> 1; o > 0; o >>= 1) {
    if ((int)threadIdx.x < o) {
      s_sum[threadIdx.x] += s_sum[threadIdx.x + o];
      s_sq[threadIdx.x]  += s_sq[threadIdx.x + o];
      s_cnt[threadIdx.x] += s_cnt[threadIdx.x + o];
    }
    __syncthreads();
  }
  if (threadIdx.x == 0) {
    stats[c] = s_sum[0];
    stats[HC + c] = s_sq[0];
    if (c == 0) stats[2 * HC] = s_cnt[0];
  }
}

__global__ void k_bn_apply(float* __restrict__ X, const float* __restrict__ gamma,
                           const float* __restrict__ beta, const float* __restrict__ stats,
                           int M, int HC)
{
  int t = blockIdx.x * blockDim.x + threadIdx.x;
  if (t >= M * HC) return;
  int c = t % HC;
  float cnt  = stats[2 * HC];
  float mean = stats[c] / cnt;
  float var  = stats[HC + c] / cnt - mean * mean;
  float v = eluf(X[t]);
  X[t] = gamma[c] * (v - mean) * rsqrtf(var + EPSbn) + beta[c];
}

// ---------------- pool5: voxel max-pool + edge coalesce + Cartesian attrs ----------------
__global__ void k_pool5_node(const float* __restrict__ h, const float* __restrict__ pos,
                             const int* __restrict__ batch, int* __restrict__ cluster,
                             int* __restrict__ cnt, float* __restrict__ possum,
                             unsigned int* __restrict__ xm)
{
  int t = blockIdx.x * blockDim.x + threadIdx.x;
  if (t >= Nn) return;
  int gx = (int)floorf(pos[3 * t]     / 16.f); gx = min(max(gx, 0), GXc - 1);
  int gy = (int)floorf(pos[3 * t + 1] / 12.f); gy = min(max(gy, 0), GYc - 1);
  int cl = batch[t] * NVOX + gx * GYc + gy;
  cluster[t] = cl;
  atomicAdd(&cnt[cl], 1);
  atomicAdd(&possum[3 * cl],     pos[3 * t]);
  atomicAdd(&possum[3 * cl + 1], pos[3 * t + 1]);
  atomicAdd(&possum[3 * cl + 2], pos[3 * t + 2]);
  for (int c = 0; c < 128; ++c)
    atomicMax(&xm[(size_t)cl * 128 + c], enc_f32(h[(size_t)t * 128 + c]));
}

__global__ void k_pool5_final(const unsigned int* __restrict__ xm, const int* __restrict__ cnt,
                              const float* __restrict__ possum, float* __restrict__ xp,
                              float* __restrict__ posp, int* __restrict__ nmask)
{
  int t = blockIdx.x * blockDim.x + threadIdx.x;
  if (t >= NCc) return;
  int ct = cnt[t];
  nmask[t] = ct > 0;
  float inv = ct > 0 ? 1.f / (float)ct : 1.f;
  for (int j = 0; j < 3; ++j) posp[3 * t + j] = possum[3 * t + j] * inv;
  for (int c = 0; c < 128; ++c) {
    float v = dec_f32(xm[(size_t)t * 128 + c]);
    xp[(size_t)t * 128 + c] = (ct > 0 && finitef(v)) ? v : 0.f;
  }
}

// edges intra-graph -> dedup key = src_cluster * NVOX + (dst_cluster % NVOX)
__global__ void k_pool5_edge_a(const int* __restrict__ cluster, const int* __restrict__ ei,
                               int* __restrict__ srcc, int* __restrict__ dstc,
                               int* __restrict__ claim)
{
  int t = blockIdx.x * blockDim.x + threadIdx.x;
  if (t >= Ee) return;
  int s = cluster[ei[t]], d = cluster[ei[Ee + t]];
  srcc[t] = s; dstc[t] = d;
  if (s != d) atomicMin(&claim[(size_t)s * NVOX + (d % NVOX)], t);  // keep lowest edge idx
}

__global__ void k_pool5_edge_b(const int* __restrict__ srcc, const int* __restrict__ dstc,
                               const int* __restrict__ claim, const float* __restrict__ posp,
                               int* __restrict__ emaskp, unsigned int* __restrict__ amax)
{
  int t = blockIdx.x * blockDim.x + threadIdx.x;
  if (t >= Ee) return;
  int s = srcc[t], d = dstc[t];
  int keep = (s != d) && (claim[(size_t)s * NVOX + (d % NVOX)] == t);
  emaskp[t] = keep;
  if (keep) {
    float mx = 0.f;
    for (int j = 0; j < 3; ++j) mx = fmaxf(mx, fabsf(posp[3 * s + j] - posp[3 * d + j]));
    atomicMax(amax, __float_as_uint(mx));   // non-negative floats: bit order == value order
  }
}

__global__ void k_pool5_eattr(const int* __restrict__ srcc, const int* __restrict__ dstc,
                              const int* __restrict__ emaskp, const float* __restrict__ posp,
                              const unsigned int* __restrict__ amax, float* __restrict__ eattrp)
{
  int t = blockIdx.x * blockDim.x + threadIdx.x;
  if (t >= Ee) return;
  int s = srcc[t], d = dstc[t];
  float inv = 0.5f / __uint_as_float(*amax);
  for (int j = 0; j < 3; ++j) {
    float cj = posp[3 * s + j] - posp[3 * d + j];
    eattrp[3 * t + j] = emaskp[t] ? (cj * inv + 0.5f) : 0.f;
  }
}

// ---------------- pool7: 4x4 voxel max over pooled nodes ----------------
__global__ void k_pool7_max(const float* __restrict__ h, const float* __restrict__ posp,
                            const int* __restrict__ nmask, unsigned int* __restrict__ z)
{
  int t = blockIdx.x * blockDim.x + threadIdx.x;
  if (t >= NCc) return;
  if (!nmask[t]) return;
  int gx = min(max((int)floorf(posp[3 * t]     / 60.f), 0), 3);
  int gy = min(max((int)floorf(posp[3 * t + 1] / 45.f), 0), 3);
  int cid = (t / NVOX) * 16 + gx * 4 + gy;
  for (int c = 0; c < 128; ++c)
    atomicMax(&z[(size_t)cid * 128 + c], enc_f32(h[(size_t)t * 128 + c]));
}

__global__ void k_pool7_fin(const unsigned int* __restrict__ z, float* __restrict__ zf)
{
  int t = blockIdx.x * blockDim.x + threadIdx.x;
  if (t >= Bg * 16 * 128) return;
  float v = dec_f32(z[t]);
  zf[t] = finitef(v) ? v : 0.f;
}

// ---------------- host orchestration ----------------
static inline int cdiv(long long a, long long b) { return (int)((a + b - 1) / b); }

extern "C" void kernel_launch(void* const* d_in, const int* in_sizes, int n_in,
                              void* d_out, int out_size, void* d_ws, size_t ws_size,
                              hipStream_t stream)
{
  (void)in_sizes; (void)n_in; (void)out_size; (void)ws_size;

  // Input order = setup_inputs() dict order; nested params flattened pytree-style
  // (alphabetical keys): x,pos,edge_index,edge_attr,batch, then
  // conv1..conv7 {We,Wk,Wq,Ws,Wv,bk,bq,bs,bv}, fc_W, norm1..norm7 {beta,gamma}.
  const float* x     = (const float*)d_in[0];
  const float* pos   = (const float*)d_in[1];
  const int*   ei    = (const int*)d_in[2];     // [2,E]: src row 0, dst row 1
  const float* eatt  = (const float*)d_in[3];
  const int*   batch = (const int*)d_in[4];
  auto convW = [&](int layer, int which) -> const float* {
    return (const float*)d_in[5 + 9 * (layer - 1) + which];
  }; // which: 0=We 1=Wk 2=Wq 3=Ws 4=Wv 5=bk 6=bq 7=bs 8=bv
  const float* fcW = (const float*)d_in[68];
  auto normP = [&](int layer, int which) -> const float* {
    return (const float*)d_in[69 + 2 * (layer - 1) + which];
  }; // which: 0=beta 1=gamma

  // ---- workspace arena ----
  char* base = (char*)d_ws;
  size_t off = 0;
  auto alloc = [&](size_t bytes) -> void* {
    void* p = base + off;
    off += (bytes + 255) & ~(size_t)255;
    return p;
  };
  float* A    = (float*)alloc((size_t)Nn * 128 * 4);   // node feature buffers
  float* Bb   = (float*)alloc((size_t)Nn * 128 * 4);
  float* Cc   = (float*)alloc((size_t)Nn * 128 * 4);
  float* Qb   = (float*)alloc((size_t)Nn * 128 * 4);
  float* Kb   = (float*)alloc((size_t)Nn * 128 * 4);
  float* Vb   = (float*)alloc((size_t)Nn * 128 * 4);
  float* alphaB = (float*)alloc((size_t)Ee * 3 * 4);
  float* aexpB  = (float*)alloc((size_t)Ee * 3 * 4);
  unsigned int* mmax = (unsigned int*)alloc((size_t)Nn * 3 * 4);
  float* denom  = (float*)alloc((size_t)Nn * 3 * 4);
  float* stats  = (float*)alloc((2 * MAXHC + 8) * 4);
  int*   cluster = (int*)alloc((size_t)Nn * 4);
  int*   cnt     = (int*)alloc((size_t)NCc * 4);
  float* possum  = (float*)alloc((size_t)NCc * 3 * 4);
  float* posp    = (float*)alloc((size_t)NCc * 3 * 4);
  int*   nmaskB  = (int*)alloc((size_t)NCc * 4);
  int*   srcc    = (int*)alloc((size_t)Ee * 4);
  int*   dstc    = (int*)alloc((size_t)Ee * 4);
  int*   emaskp  = (int*)alloc((size_t)Ee * 4);
  int*   claim   = (int*)alloc((size_t)NCc * NVOX * 4);   // per-graph (src,dst) table
  float* eattrp  = (float*)alloc((size_t)Ee * 3 * 4);
  unsigned int* amax = (unsigned int*)alloc(256);
  unsigned int* xm   = (unsigned int*)alloc((size_t)NCc * 128 * 4);
  unsigned int* zenc = (unsigned int*)alloc((size_t)Bg * 16 * 128 * 4);
  float* zf          = (float*)alloc((size_t)Bg * 16 * 128 * 4);

  auto fill_f = [&](float* p, float v, size_t n) {
    k_fill_f<<<cdiv((long long)n, 256), 256, 0, stream>>>(p, v, n);
  };
  auto fill_u = [&](unsigned int* p, unsigned int v, size_t n) {
    k_fill_u<<<cdiv((long long)n, 256), 256, 0, stream>>>(p, v, n);
  };
  auto fill_i = [&](int* p, int v, size_t n) {
    k_fill_i<<<cdiv((long long)n, 256), 256, 0, stream>>>(p, v, n);
  };

  auto gemm = [&](const float* X, const float* W, const float* bias, float* Y,
                  int M, int Kd, int Out) {
    if (Kd >= 32 && (Kd % 4) == 0 && (Out % 16) == 0 && (M % 16) == 0) {
      dim3 grid(cdiv(M, 32), cdiv(Out, 64));  // 4 waves/block, 32x64 tile
      k_gemm_wmma<<<grid, 128, 0, stream>>>(X, W, bias, Y, M, Kd, Out);
    } else {
      long long tot = (long long)M * Out;
      k_gemm_scalar<<<cdiv(tot, 256), 256, 0, stream>>>(X, W, bias, Y, M, Kd, Out);
    }
  };

  // Full TransformerConv layer + ELU + BatchNorm (in-place on Yout)
  auto conv = [&](int layer, const float* Xin, float* Yout, int M, int fin, int H, int C,
                  const int* srcP, const int* dstP, const float* eaP, const int* emP) {
    const int HC = H * C;
    gemm(Xin, convW(layer, 2), convW(layer, 6), Qb,   M, fin, HC);  // Wq,bq
    gemm(Xin, convW(layer, 1), convW(layer, 5), Kb,   M, fin, HC);  // Wk,bk
    gemm(Xin, convW(layer, 4), convW(layer, 8), Vb,   M, fin, HC);  // Wv,bv
    gemm(Xin, convW(layer, 3), convW(layer, 7), Yout, M, fin, HC);  // Ws,bs (root term)
    fill_u(mmax, ENC_NEG_INF, (size_t)M * H);
    fill_f(denom, 0.f, (size_t)M * H);
    long long TH = (long long)Ee * H;
    k_edge_alpha<<<cdiv(TH, 256), 256, 0, stream>>>(Qb, Kb, convW(layer, 0), eaP,
                                                    srcP, dstP, emP, alphaB, mmax, Ee, H, C);
    k_edge_softmax<<<cdiv(TH, 256), 256, 0, stream>>>(alphaB, mmax, dstP, emP,
                                                      aexpB, denom, Ee, H);
    k_edge_scatter<<<cdiv(TH, 256), 256, 0, stream>>>(Vb, convW(layer, 0), eaP,
                                                      srcP, dstP, emP, aexpB, denom,
                                                      Yout, Ee, H, C);
    const int* nm = (layer >= 6) ? nmaskB : nullptr;
    k_bn_stats<<<HC, 256, 0, stream>>>(Yout, nm, stats, M, HC);
    k_bn_apply<<<cdiv((long long)M * HC, 256), 256, 0, stream>>>(
        Yout, normP(layer, 1), normP(layer, 0), stats, M, HC);
  };

  const int* src0 = ei;
  const int* dst0 = ei + Ee;

  // ---- network ----
  conv(1, x,  A,  Nn, 1,   3, 16,  src0, dst0, eatt, nullptr);
  conv(2, A,  Bb, Nn, 48,  1, 32,  src0, dst0, eatt, nullptr);   // sc = Bb
  conv(3, Bb, A,  Nn, 32,  3, 32,  src0, dst0, eatt, nullptr);
  conv(4, A,  Cc, Nn, 96,  1, 32,  src0, dst0, eatt, nullptr);
  k_add<<<cdiv((long long)Nn * 32, 256), 256, 0, stream>>>(Cc, Bb, (size_t)Nn * 32);
  conv(5, Cc, A,  Nn, 32,  1, 128, src0, dst0, eatt, nullptr);

  // pool5: A[N,128] -> Bb[NC,128]
  fill_i(cnt, 0, NCc);
  fill_f(possum, 0.f, (size_t)NCc * 3);
  fill_u(xm, ENC_NEG_INF, (size_t)NCc * 128);
  k_pool5_node<<<cdiv(Nn, 256), 256, 0, stream>>>(A, pos, batch, cluster, cnt, possum, xm);
  k_pool5_final<<<cdiv(NCc, 256), 256, 0, stream>>>(xm, cnt, possum, Bb, posp, nmaskB);
  fill_i(claim, 0x7FFFFFFF, (size_t)NCc * NVOX);
  k_pool5_edge_a<<<cdiv(Ee, 256), 256, 0, stream>>>(cluster, ei, srcc, dstc, claim);
  fill_u(amax, 0u, 1);
  k_pool5_edge_b<<<cdiv(Ee, 256), 256, 0, stream>>>(srcc, dstc, claim, posp, emaskp, amax);
  k_pool5_eattr<<<cdiv(Ee, 256), 256, 0, stream>>>(srcc, dstc, emaskp, posp, amax, eattrp);

  conv(6, Bb, A,  NCc, 128, 3, 128, srcc, dstc, eattrp, emaskp);   // sc = Bb (pooled)
  conv(7, A,  Cc, NCc, 384, 1, 128, srcc, dstc, eattrp, emaskp);
  k_add<<<cdiv((long long)NCc * 128, 256), 256, 0, stream>>>(Cc, Bb, (size_t)NCc * 128);

  // pool7 + FC
  fill_u(zenc, ENC_NEG_INF, (size_t)Bg * 16 * 128);
  k_pool7_max<<<cdiv(NCc, 256), 256, 0, stream>>>(Cc, posp, nmaskB, zenc);
  k_pool7_fin<<<cdiv(Bg * 16 * 128, 256), 256, 0, stream>>>(zenc, zf);
  gemm(zf, fcW, nullptr, (float*)d_out, Bg, 16 * 128, NOUTc);  // [16,2048]@[2048,101]
}
